// Net_42614665510980
// MI455X (gfx1250) — matrix operand order repaired and checked by reading, not compiled, not run on hardware
//
#include <hip/hip_runtime.h>
#include <hip/hip_bf16.h>
#include <utility>

// ---------------------------------------------------------------------------
// Types for CDNA5 WMMA (wave32): v_wmma_f32_16x16x32_bf16
// ---------------------------------------------------------------------------
typedef __bf16 bf16_t;
typedef __attribute__((ext_vector_type(16))) __bf16 bf16x16;
typedef __attribute__((ext_vector_type(8)))  float  f32x8;
typedef __attribute__((ext_vector_type(4)))  unsigned int u32x4;
typedef __attribute__((vector_size(16)))     int    i32x4v;

#define TSEQ 64
#define BATCH 32
#define HDIM 1024
#define GDIM 4096       // 4*H
#define DPAD 320        // 300 padded to multiple of 32
#define DRAW 300
#define VOCAB 32000
#define KC 64           // K-chunk staged into LDS per GEMM iteration

// ---------------------------------------------------------------------------
// Async Global->LDS path (gfx1250). Guarded: falls back to synchronous copy.
// Builtin signature (from compiler diagnostic): first param is
//   'int __attribute__((vector_size(16))) __device__ *'  (AS1 int4*)
// ---------------------------------------------------------------------------
#if __has_builtin(__builtin_amdgcn_global_load_async_to_lds_b128) && \
    __has_builtin(__builtin_amdgcn_s_wait_asynccnt)
#define HAVE_ASYNC_LDS 1
#else
#define HAVE_ASYNC_LDS 0
#endif

typedef __attribute__((address_space(1))) i32x4v as1_i32x4;
typedef __attribute__((address_space(3))) i32x4v as3_i32x4;

__device__ __forceinline__ void cp_b128_to_lds(const bf16_t* g, bf16_t* l) {
#if HAVE_ASYNC_LDS
    __builtin_amdgcn_global_load_async_to_lds_b128(
        (as1_i32x4*)(g), (as3_i32x4*)(l), 0, 0);
#else
    *reinterpret_cast<u32x4*>(l) = *reinterpret_cast<const u32x4*>(g);
#endif
}

__device__ __forceinline__ void wait_async_lds() {
#if HAVE_ASYNC_LDS
    __builtin_amdgcn_s_wait_asynccnt(0);
#endif
}

// ---------------------------------------------------------------------------
// Fragment loaders (ISA 7.12.2 layouts)
// A 16x32 bf16: lane L<16 holds row M=L, K = k0+{0..7} and k0+{16..23};
//               lane L>=16 holds row M=L-16, K = k0+{8..15} and k0+{24..31}.
// B 32x16 bf16: lane L<16 holds col N=L, K = k0+{0..7} and k0+{16..23};
//               lane L>=16 holds col N=L-16, K = k0+{8..15} and k0+{24..31}.
// C   16x16 f32: VGPR i holds (M = i + 8*(lane>=16), N = lane&15).
// ---------------------------------------------------------------------------
__device__ __forceinline__ bf16x16 load_frag_a(const bf16_t* __restrict__ base,
                                               int ld, int m0, int k0, int lane) {
    int row  = m0 + (lane & 15);
    int koff = k0 + ((lane >> 4) << 3);
    const bf16_t* p = base + row * ld + koff;
    bf16x16 f;
    reinterpret_cast<u32x4*>(&f)[0] = *reinterpret_cast<const u32x4*>(p);
    reinterpret_cast<u32x4*>(&f)[1] = *reinterpret_cast<const u32x4*>(p + 16);
    return f;
}

__device__ __forceinline__ bf16x16 load_frag_b(const bf16_t* w,
                                               int ld, int n0, int k0, int lane) {
    int col  = n0 + (lane & 15);
    int koff = k0 + ((lane >> 4) << 3);
    const bf16_t* p = w + col * ld + koff;
    bf16x16 f;
    reinterpret_cast<u32x4*>(&f)[0] = *reinterpret_cast<const u32x4*>(p);
    reinterpret_cast<u32x4*>(&f)[1] = *reinterpret_cast<const u32x4*>(p + 16);
    return f;
}

__device__ __forceinline__ f32x8 wmma_bf16(const bf16x16& a, const bf16x16& b, const f32x8& c) {
    return __builtin_amdgcn_wmma_f32_16x16x32_bf16(false, a, false, b, (short)0, c, false, false);
}

__device__ __forceinline__ float sigmoidf_(float x) { return 1.0f / (1.0f + __expf(-x)); }

// ---------------------------------------------------------------------------
// Generic GEMM: C[M,N] = A[M,K](bf16) * W[N,K]^T(bf16) + b1[n] (+ b2[n])
// Workgroup = 256 threads = 8 waves; wave tile 32x64; WG tile 64x256.
// B tile (256 x KC) is double-buffered in LDS via async Global->LDS copies,
// overlapped with WMMA on the other buffer.  A fragments stay in VMEM (tiny,
// L1/L2 resident).  grid = (N/256, M/64).
// mode 0: C[m*N+n]; mode 1: logits permute [B,T,V].
// ---------------------------------------------------------------------------
__global__ __launch_bounds__(256) void gemm_bf16_kernel(
    const bf16_t* __restrict__ A, int lda,
    const bf16_t* __restrict__ W, int ldw,
    const float* __restrict__ b1, const float* __restrict__ b2,
    float* __restrict__ C, int N, int K, int mode)
{
    __shared__ bf16_t Bs[2][256 * KC];     // 2 x 32 KB

    const int tid  = threadIdx.x;
    const int w    = tid >> 5;
    const int lane = tid & 31;
    const int wm = w >> 2, wn = w & 3;
    const int mblk = blockIdx.y * 64;
    const int nblk = blockIdx.x * 256;
    const int m0 = mblk + wm * 32;
    const int nw = wn * 64;                 // wave's column offset inside LDS tile

    f32x8 acc[2][4];
#pragma unroll
    for (int s = 0; s < 4; ++s) {
        int n = nblk + nw + s * 16 + (lane & 15);
        float bv = b1[n] + (b2 ? b2[n] : 0.0f);
#pragma unroll
        for (int mt = 0; mt < 2; ++mt)
#pragma unroll
            for (int i = 0; i < 8; ++i) acc[mt][s][i] = bv;
    }

    const int nk = K / KC;

    // stage chunk 0: 256 rows x KC elems; 8 x b128 per thread
    for (int u = tid; u < 256 * (KC / 8); u += 256) {
        int r = u >> 3, ck = (u & 7) << 3;
        cp_b128_to_lds(W + (size_t)(nblk + r) * ldw + ck, &Bs[0][r * KC + ck]);
    }

    int cur = 0;
    for (int kc = 0; kc < nk; ++kc) {
        wait_async_lds();
        __syncthreads();                    // buffer `cur` ready for everyone

        if (kc + 1 < nk) {                  // prefetch next chunk into other buffer
            int k0n = (kc + 1) * KC;
            for (int u = tid; u < 256 * (KC / 8); u += 256) {
                int r = u >> 3, ck = (u & 7) << 3;
                cp_b128_to_lds(W + (size_t)(nblk + r) * ldw + k0n + ck,
                               &Bs[cur ^ 1][r * KC + ck]);
            }
        }

#pragma unroll
        for (int kk = 0; kk < KC; kk += 32) {
            int kg = kc * KC + kk;
            if (kg + 64 < K)
                __builtin_prefetch(A + (m0 + (lane & 15)) * lda + kg + 64, 0, 3);
            bf16x16 a0 = load_frag_a(A, lda, m0,      kg, lane);
            bf16x16 a1 = load_frag_a(A, lda, m0 + 16, kg, lane);
#pragma unroll
            for (int s = 0; s < 4; ++s) {
                bf16x16 bb = load_frag_b(&Bs[cur][0], KC, nw + s * 16, kk, lane);
                acc[0][s] = wmma_bf16(a0, bb, acc[0][s]);
                acc[1][s] = wmma_bf16(a1, bb, acc[1][s]);
            }
        }
        __syncthreads();                    // done reading `cur` before reuse
        cur ^= 1;
    }

    const int nlo   = lane & 15;
    const int mbase = (lane >> 4) << 3;
#pragma unroll
    for (int mt = 0; mt < 2; ++mt) {
#pragma unroll
        for (int s = 0; s < 4; ++s) {
            int n = nblk + nw + s * 16 + nlo;
#pragma unroll
            for (int i = 0; i < 8; ++i) {
                int m = m0 + mt * 16 + mbase + i;
                if (mode == 0) {
                    C[m * N + n] = acc[mt][s][i];
                } else {
                    int t = m >> 5, b = m & 31;            // row = t*B + b
                    C[(b * TSEQ + t) * VOCAB + n] = acc[mt][s][i];
                }
            }
        }
    }
}

// ---------------------------------------------------------------------------
// One LSTM recurrence step.  gates = gIn(t) + h_in @ Whh^T, then cell update.
// 32 workgroups, each owning a 32-wide band j0..j0+31 of all 4 gates, so the
// cell update is workgroup-local (LDS) — no grid sync inside a step.
// h is double-buffered between step launches.  Whh is L2-resident bf16, read
// directly (latency-critical path; no extra barriers).
// ---------------------------------------------------------------------------
__global__ __launch_bounds__(256) void lstm_step_kernel(
    const bf16_t* __restrict__ Whh,      // [4096,1024] bf16
    const float*  __restrict__ gIn,      // [32,4096] input-side gates + bias (step t)
    const bf16_t* __restrict__ h_in,     // [32,1024] bf16
    float*  __restrict__ cbuf,           // [32,1024] f32 (band-exclusive)
    bf16_t* __restrict__ h_out,          // [32,1024] bf16
    float*  __restrict__ h_out_f,        // [32,1024] f32 (final-state capture)
    bf16_t* __restrict__ hs_out)         // [32,1024] bf16 (sequence store, step t)
{
    __shared__ float lds_g[32 * 4 * 32];  // [m][gate][jc] = 16 KB

    const int j0   = blockIdx.x * 32;
    const int w    = threadIdx.x >> 5;
    const int lane = threadIdx.x & 31;
    const int G  = w >> 1;          // gate 0..3 (i,f,g,o)
    const int m0 = (w & 1) * 16;    // M tile
    const int nlo   = lane & 15;
    const int mbase = (lane >> 4) << 3;

    f32x8 acc[2];
#pragma unroll
    for (int s = 0; s < 2; ++s) {
        int n = G * HDIM + j0 + s * 16 + nlo;
#pragma unroll
        for (int i = 0; i < 8; ++i)
            acc[s][i] = gIn[(m0 + mbase + i) * GDIM + n];
    }

    for (int k0 = 0; k0 < HDIM; k0 += 32) {
        bf16x16 a = load_frag_a(h_in, HDIM, m0, k0, lane);
#pragma unroll
        for (int s = 0; s < 2; ++s) {
            bf16x16 bb = load_frag_b(Whh, HDIM, G * HDIM + j0 + s * 16, k0, lane);
            acc[s] = wmma_bf16(a, bb, acc[s]);
        }
    }

#pragma unroll
    for (int s = 0; s < 2; ++s) {
        int jc = s * 16 + nlo;
#pragma unroll
        for (int i = 0; i < 8; ++i) {
            int m = m0 + mbase + i;
            lds_g[(m * 4 + G) * 32 + jc] = acc[s][i];
        }
    }
    __syncthreads();

    for (int e = threadIdx.x; e < 32 * 32; e += 256) {
        int m = e >> 5, jc = e & 31;
        const float* g = &lds_g[m * 4 * 32 + jc];
        float gi = g[0], gf = g[32], gg = g[64], go = g[96];
        int idx = m * HDIM + j0 + jc;
        float c  = cbuf[idx];
        float cn = sigmoidf_(gf) * c + sigmoidf_(gi) * tanhf(gg);
        float hn = sigmoidf_(go) * tanhf(cn);
        cbuf[idx]    = cn;
        h_out_f[idx] = hn;
        bf16_t hb = (bf16_t)hn;
        h_out[idx]  = hb;
        hs_out[idx] = hb;
    }
}

// ---------------------------------------------------------------------------
// Small utility kernels
// ---------------------------------------------------------------------------
__global__ void cvt_pad_kernel(const float* __restrict__ src, bf16_t* __restrict__ dst,
                               int rows, int sk, int dk)
{
    long total = (long)rows * dk;
    for (long i = blockIdx.x * (long)blockDim.x + threadIdx.x; i < total;
         i += (long)gridDim.x * blockDim.x) {
        int r = (int)(i / dk), c = (int)(i % dk);
        dst[i] = (c < sk) ? (bf16_t)src[(long)r * sk + c] : (bf16_t)0.0f;
    }
}

__global__ void embed_kernel(const float* __restrict__ emb, const int* __restrict__ word,
                             bf16_t* __restrict__ encx, bf16_t* __restrict__ decx)
{
    int idx = blockIdx.x * blockDim.x + threadIdx.x;
    const int per = TSEQ * BATCH * DPAD;
    if (idx >= 2 * per) return;
    int half = idx / per;
    int rem  = idx % per;
    int r = rem / DPAD;      // t*B + b
    int d = rem % DPAD;
    int t = r >> 5, b = r & 31;
    int tok = (half == 0) ? word[b * TSEQ + t]
                          : ((t == 0) ? 1 : word[b * TSEQ + (t - 1)]);   // BOS = 1
    float v = (d < DRAW) ? emb[tok * DRAW + d] : 0.0f;
    bf16_t o = (bf16_t)v;
    if (half == 0) encx[r * DPAD + d] = o;
    else           decx[r * DPAD + d] = o;
}

__global__ void zero_f32_kernel(float* p, int n)
{
    for (int i = blockIdx.x * blockDim.x + threadIdx.x; i < n; i += gridDim.x * blockDim.x)
        p[i] = 0.0f;
}

__global__ void mix_state_kernel(const float* __restrict__ hf0, const float* __restrict__ hf1,
                                 const float* __restrict__ c0,  const float* __restrict__ c1,
                                 bf16_t* __restrict__ hdec, float* __restrict__ cd, int n)
{
    for (int i = blockIdx.x * blockDim.x + threadIdx.x; i < n; i += gridDim.x * blockDim.x) {
        cd[i] = 0.5f * (c0[i] + c1[i]);
        float hv = 0.5f * (hf0[i] + hf1[i]);
        hdec[i] = (bf16_t)hv;
    }
}

// ---------------------------------------------------------------------------
// Host orchestration
// ---------------------------------------------------------------------------
static inline char* bump(char*& p, size_t bytes) {
    char* r = p;
    p += (bytes + 255) & ~(size_t)255;
    return r;
}

extern "C" void kernel_launch(void* const* d_in, const int* in_sizes, int n_in,
                              void* d_out, int out_size, void* d_ws, size_t ws_size,
                              hipStream_t stream)
{
    const float* emb      = (const float*)d_in[0];
    const float* eWih0    = (const float*)d_in[1];
    const float* eWhh0    = (const float*)d_in[2];
    const float* eBih0    = (const float*)d_in[3];
    const float* eBhh0    = (const float*)d_in[4];
    const float* eWih1    = (const float*)d_in[5];
    const float* eWhh1    = (const float*)d_in[6];
    const float* eBih1    = (const float*)d_in[7];
    const float* eBhh1    = (const float*)d_in[8];
    const float* dWih     = (const float*)d_in[9];
    const float* dWhh     = (const float*)d_in[10];
    const float* dBih     = (const float*)d_in[11];
    const float* dBhh     = (const float*)d_in[12];
    const float* outW     = (const float*)d_in[13];
    const float* outB     = (const float*)d_in[14];
    const int*   word     = (const int*)d_in[15];
    float* out = (float*)d_out;

    char* ws = (char*)d_ws;
    bf16_t* wih0p = (bf16_t*)bump(ws, (size_t)GDIM * DPAD * 2);
    bf16_t* wih1b = (bf16_t*)bump(ws, (size_t)GDIM * HDIM * 2);
    bf16_t* wdecp = (bf16_t*)bump(ws, (size_t)GDIM * DPAD * 2);
    bf16_t* whh0b = (bf16_t*)bump(ws, (size_t)GDIM * HDIM * 2);
    bf16_t* whh1b = (bf16_t*)bump(ws, (size_t)GDIM * HDIM * 2);
    bf16_t* whhdb = (bf16_t*)bump(ws, (size_t)GDIM * HDIM * 2);
    bf16_t* outwb = (bf16_t*)bump(ws, (size_t)VOCAB * HDIM * 2);
    bf16_t* encx  = (bf16_t*)bump(ws, (size_t)TSEQ * BATCH * DPAD * 2);
    bf16_t* decx  = (bf16_t*)bump(ws, (size_t)TSEQ * BATCH * DPAD * 2);
    float*  gIn   = (float*) bump(ws, (size_t)TSEQ * BATCH * GDIM * 4);
    bf16_t* hsA   = (bf16_t*)bump(ws, (size_t)TSEQ * BATCH * HDIM * 2);
    bf16_t* hsD   = (bf16_t*)bump(ws, (size_t)TSEQ * BATCH * HDIM * 2);
    bf16_t* hping = (bf16_t*)bump(ws, (size_t)BATCH * HDIM * 2);
    bf16_t* hpong = (bf16_t*)bump(ws, (size_t)BATCH * HDIM * 2);
    float*  hf0   = (float*) bump(ws, (size_t)BATCH * HDIM * 4);
    float*  hf1   = (float*) bump(ws, (size_t)BATCH * HDIM * 4);
    float*  c0b   = (float*) bump(ws, (size_t)BATCH * HDIM * 4);
    float*  c1b   = (float*) bump(ws, (size_t)BATCH * HDIM * 4);
    float*  cdb   = (float*) bump(ws, (size_t)BATCH * HDIM * 4);

    // --- weight conversion to bf16 (K-padded for D=300 matrices) ---
    cvt_pad_kernel<<<1024, 256, 0, stream>>>(eWih0, wih0p, GDIM, DRAW, DPAD);
    cvt_pad_kernel<<<1024, 256, 0, stream>>>(eWih1, wih1b, GDIM, HDIM, HDIM);
    cvt_pad_kernel<<<1024, 256, 0, stream>>>(dWih,  wdecp, GDIM, DRAW, DPAD);
    cvt_pad_kernel<<<1024, 256, 0, stream>>>(eWhh0, whh0b, GDIM, HDIM, HDIM);
    cvt_pad_kernel<<<1024, 256, 0, stream>>>(eWhh1, whh1b, GDIM, HDIM, HDIM);
    cvt_pad_kernel<<<1024, 256, 0, stream>>>(dWhh,  whhdb, GDIM, HDIM, HDIM);
    cvt_pad_kernel<<<2048, 256, 0, stream>>>(outW,  outwb, VOCAB, HDIM, HDIM);

    // --- embeddings (encoder tokens + teacher-forced decoder tokens) ---
    embed_kernel<<<(2 * TSEQ * BATCH * DPAD + 255) / 256, 256, 0, stream>>>(emb, word, encx, decx);

    const int M  = TSEQ * BATCH;       // 2048
    const int nHC = BATCH * HDIM;      // 32768

    // ================= encoder layer 0 =================
    zero_f32_kernel<<<64, 256, 0, stream>>>(c0b, nHC);
    zero_f32_kernel<<<64, 256, 0, stream>>>((float*)hping, nHC / 2);
    gemm_bf16_kernel<<<dim3(GDIM / 256, M / 64), 256, 0, stream>>>(
        encx, DPAD, wih0p, DPAD, eBih0, eBhh0, gIn, GDIM, DPAD, 0);
    {
        bf16_t* hin = hping; bf16_t* hout = hpong;
        for (int t = 0; t < TSEQ; ++t) {
            lstm_step_kernel<<<32, 256, 0, stream>>>(
                whh0b, gIn + (size_t)t * BATCH * GDIM, hin, c0b, hout, hf0,
                hsA + (size_t)t * BATCH * HDIM);
            std::swap(hin, hout);
        }
    }

    // ================= encoder layer 1 =================
    zero_f32_kernel<<<64, 256, 0, stream>>>(c1b, nHC);
    zero_f32_kernel<<<64, 256, 0, stream>>>((float*)hping, nHC / 2);
    gemm_bf16_kernel<<<dim3(GDIM / 256, M / 64), 256, 0, stream>>>(
        hsA, HDIM, wih1b, HDIM, eBih1, eBhh1, gIn, GDIM, HDIM, 0);
    {
        bf16_t* hin = hping; bf16_t* hout = hpong;
        for (int t = 0; t < TSEQ; ++t) {
            lstm_step_kernel<<<32, 256, 0, stream>>>(
                whh1b, gIn + (size_t)t * BATCH * GDIM, hin, c1b, hout, hf1,
                hsA + (size_t)t * BATCH * HDIM);   // dummy hs (hsA already consumed)
            std::swap(hin, hout);
        }
    }

    // ================= decoder init: mean over layers =================
    mix_state_kernel<<<64, 256, 0, stream>>>(hf0, hf1, c0b, c1b, hping, cdb, nHC);

    // ================= decoder =================
    gemm_bf16_kernel<<<dim3(GDIM / 256, M / 64), 256, 0, stream>>>(
        decx, DPAD, wdecp, DPAD, dBih, dBhh, gIn, GDIM, DPAD, 0);
    {
        bf16_t* hin = hping; bf16_t* hout = hpong;
        for (int t = 0; t < TSEQ; ++t) {
            lstm_step_kernel<<<32, 256, 0, stream>>>(
                whhdb, gIn + (size_t)t * BATCH * GDIM, hin, cdb, hout, hf0,
                hsD + (size_t)t * BATCH * HDIM);
            std::swap(hin, hout);
        }
    }

    // ================= output projection -> logits [B,T,V] =================
    gemm_bf16_kernel<<<dim3(VOCAB / 256, M / 64), 256, 0, stream>>>(
        hsD, HDIM, outwb, HDIM, outB, nullptr, out, VOCAB, HDIM, 1);
}
